// SpatialRefine_78838419685474
// MI455X (gfx1250) — matrix operand order repaired and checked by reading, not compiled, not run on hardware
//
#include <hip/hip_runtime.h>
#include <hip/hip_bf16.h>
#include <math.h>

typedef __attribute__((ext_vector_type(16))) _Float16 v16h;
typedef __attribute__((ext_vector_type(8)))  _Float16 v8h;
typedef __attribute__((ext_vector_type(8)))  float    v8f;
typedef __attribute__((ext_vector_type(4)))  float    v4f;

// ---------------------------------------------------------------------------
// WMMA fragment loaders from LDS, matching CDNA5 16x16x32 f16 layouts.
// Source arrays are row-major [rows][32] _Float16, 16B-aligned rows (64B).
// A (16x32): lane<16 -> K{0..7} then K{16..23}; lane>=16 -> K{8..15},K{24..31}
// B (32x16): lane<16 -> K{0..7} then K{16..23}; lane>=16 -> K{8..15},K{24..31}
// ---------------------------------------------------------------------------
__device__ __forceinline__ v16h cat16(v8h lo, v8h hi) {
  v16h r;
#pragma unroll
  for (int i = 0; i < 8; ++i) { r[i] = lo[i]; r[i + 8] = hi[i]; }
  return r;
}

__device__ __forceinline__ v16h load_fragA(const _Float16* base, int row, int laneHi) {
  const _Float16* r = base + row * 32;
  v8h lo = *(const v8h*)(r + (laneHi ? 8  : 0));
  v8h hi = *(const v8h*)(r + (laneHi ? 24 : 16));
  return cat16(lo, hi);
}

__device__ __forceinline__ v16h load_fragB(const _Float16* base, int col, int laneHi) {
  const _Float16* r = base + col * 32;
  v8h lo = *(const v8h*)(r + (laneHi ? 8  : 0));
  v8h hi = *(const v8h*)(r + (laneHi ? 24 : 16));
  return cat16(lo, hi);
}

// ---------------------------------------------------------------------------
// Fused SpatialRefine kernel. One 256-thread block (8 wave32) per 16x16 tile.
// ---------------------------------------------------------------------------
__launch_bounds__(256)
__global__ void spatial_refine_fused(const float* __restrict__ flow,
                                     const float* __restrict__ w1,
                                     const float* __restrict__ b1,
                                     const float* __restrict__ w2,
                                     const float* __restrict__ b2,
                                     float* __restrict__ out,
                                     int H, int W)
{
  __shared__ __align__(16) float    flowS[2][22][22];   // flow halo (+/-3), zero padded
  __shared__ __align__(16) _Float16 guidS[18][18][4];   // guidance for conv halo (+/-1)
  __shared__ __align__(16) _Float16 B1[256][32];        // im2col: [pixel][k], k=tap*3+c
  __shared__ __align__(16) _Float16 hS[256][32];        // hidden:  [pixel][ch]
  __shared__ __align__(16) _Float16 W1s[32][32];        // conv3x3 weights, K padded 27->32
  __shared__ __align__(16) _Float16 W2s[64][32];        // conv1x1 weights, M padded 50->64
  __shared__ __align__(16) float    kernS[256][64];     // per-pixel 50 kernels (f32)
  __shared__ float b1s[32];
  __shared__ float b2s[64];

  const int tid    = threadIdx.x;
  const int lane   = tid & 31;
  const int wv     = tid >> 5;
  const int laneHi = lane >> 4;        // 0 for lanes 0-15, 1 for lanes 16-31
  const int N      = lane & 15;        // WMMA column / A-row-within-tile index

  const int x0 = blockIdx.x * 16;
  const int y0 = blockIdx.y * 16;
  const int b  = blockIdx.z;
  const size_t planeHW = (size_t)H * (size_t)W;
  const float* flowB = flow + (size_t)b * 2 * planeHW;

  // ---- Stage A: stage weights/biases and flow halo into LDS ---------------
  for (int idx = tid; idx < 32 * 32; idx += 256) {
    int m = idx >> 5, k = idx & 31;
    float v = 0.f;
    if (k < 27) { int tap = k / 3, c = k - tap * 3; v = w1[m * 27 + c * 9 + tap]; }
    W1s[m][k] = (_Float16)v;
  }
  for (int idx = tid; idx < 64 * 32; idx += 256) {
    int m = idx >> 5, k = idx & 31;
    W2s[m][k] = (_Float16)((m < 50) ? w2[m * 32 + k] : 0.f);
  }
  if (tid < 32) b1s[tid] = b1[tid];
  if (tid < 64) b2s[tid] = (tid < 50) ? b2[tid] : 0.f;

  for (int idx = tid; idx < 2 * 22 * 22; idx += 256) {
    int c = idx / 484;
    int r = (idx - c * 484) / 22;
    int q = idx - c * 484 - r * 22;
    int gy = y0 - 3 + r, gx = x0 - 3 + q;
    float v = 0.f;
    if (gy >= 0 && gy < H && gx >= 0 && gx < W)
      v = flowB[(size_t)c * planeHW + (size_t)gy * W + gx];
    flowS[c][r][q] = v;
  }
  __syncthreads();

  // ---- Stage B: guidance = [flow_x, flow_y, exp(-|var|)] on 18x18 halo ----
  for (int idx = tid; idx < 18 * 18; idx += 256) {
    int py = idx / 18, px = idx - py * 18;
    int gy = y0 - 1 + py, gx = x0 - 1 + px;
    _Float16 g0 = (_Float16)0.f, g1 = (_Float16)0.f, g2 = (_Float16)0.f;
    if (gy >= 0 && gy < H && gx >= 0 && gx < W) {
      float var2 = 0.f;
#pragma unroll
      for (int c = 0; c < 2; ++c) {
        float s1 = 0.f, s2 = 0.f;
#pragma unroll
        for (int dy = 0; dy < 5; ++dy)
#pragma unroll
          for (int dx = 0; dx < 5; ++dx) {
            float v = flowS[c][py + dy][px + dx];
            s1 += v; s2 += v * v;
          }
        float var = (s2 - s1 * s1 * (1.f / 25.f)) * (1.f / 24.f);
        var2 += var * var;
      }
      float cons = expf(-sqrtf(var2));
      g0 = (_Float16)flowS[0][py + 2][px + 2];
      g1 = (_Float16)flowS[1][py + 2][px + 2];
      g2 = (_Float16)cons;
    }
    guidS[py][px][0] = g0;
    guidS[py][px][1] = g1;
    guidS[py][px][2] = g2;
    guidS[py][px][3] = (_Float16)0.f;
  }
  __syncthreads();

  // ---- Stage C: per-pixel im2col row for conv3x3, K-contiguous ------------
  {
    int p = tid, ly = p >> 4, lx = p & 15;
    int k = 0;
#pragma unroll
    for (int ty = 0; ty < 3; ++ty)
#pragma unroll
      for (int tx = 0; tx < 3; ++tx) {
        // guidance grid origin is (y0-1, x0-1): window pos = ly+ty, lx+tx
        B1[p][k + 0] = guidS[ly + ty][lx + tx][0];
        B1[p][k + 1] = guidS[ly + ty][lx + tx][1];
        B1[p][k + 2] = guidS[ly + ty][lx + tx][2];
        k += 3;
      }
#pragma unroll
    for (int z = 27; z < 32; ++z) B1[p][z] = (_Float16)0.f;
  }
  __syncthreads();

  // ---- Stage D: conv3x3 via WMMA (2 px-tiles x 2 row-tiles per wave) ------
#pragma unroll
  for (int t = 0; t < 2; ++t) {
    const int px0 = 32 * wv + 16 * t;
    v16h bf = load_fragB(&B1[0][0], px0 + N, laneHi);
#pragma unroll
    for (int r = 0; r < 2; ++r) {
      v16h af = load_fragA(&W1s[0][0], 16 * r + N, laneHi);
      v8f c = {};
      c = __builtin_amdgcn_wmma_f32_16x16x32_f16(false, af, false, bf,
                                                 (short)0, c, false, false);
      const int mbase = 16 * r + 8 * laneHi;
      v8h hv;
#pragma unroll
      for (int e = 0; e < 8; ++e)
        hv[e] = (_Float16)fmaxf(c[e] + b1s[mbase + e], 0.f);
      *(v8h*)&hS[px0 + N][mbase] = hv;   // one ds_store_b128
    }
  }

  // ---- Stage E: conv1x1 (32 -> 50, padded 64) via WMMA --------------------
#pragma unroll
  for (int t = 0; t < 2; ++t) {
    const int px0 = 32 * wv + 16 * t;
    v16h bf = load_fragB(&hS[0][0], px0 + N, laneHi);
#pragma unroll
    for (int r = 0; r < 4; ++r) {
      v16h af = load_fragA(&W2s[0][0], 16 * r + N, laneHi);
      v8f c = {};
      c = __builtin_amdgcn_wmma_f32_16x16x32_f16(false, af, false, bf,
                                                 (short)0, c, false, false);
      const int mbase = 16 * r + 8 * laneHi;
      v4f lo4, hi4;
#pragma unroll
      for (int e = 0; e < 4; ++e) {
        lo4[e] = c[e]     + b2s[mbase + e];
        hi4[e] = c[e + 4] + b2s[mbase + 4 + e];
      }
      *(v4f*)&kernS[px0 + N][mbase]     = lo4;
      *(v4f*)&kernS[px0 + N][mbase + 4] = hi4;
    }
  }
  __syncthreads();

  // ---- Stage F: refined[c] = sum_tap kernels[c*25+tap] * patch ------------
  {
    int p = tid, ly = p >> 4, lx = p & 15;
    float acc0 = 0.f, acc1 = 0.f;
#pragma unroll
    for (int dy = 0; dy < 5; ++dy)
#pragma unroll
      for (int dx = 0; dx < 5; ++dx) {
        int tap = dy * 5 + dx;
        // patch = flow[y+dy-2, x+dx-2]; flowS origin is (y0-3, x0-3)
        acc0 += kernS[p][tap]      * flowS[0][ly + 1 + dy][lx + 1 + dx];
        acc1 += kernS[p][25 + tap] * flowS[1][ly + 1 + dy][lx + 1 + dx];
      }
    int gy = y0 + ly, gx = x0 + lx;
    out[(size_t)b * 2 * planeHW +            (size_t)gy * W + gx] = acc0;
    out[(size_t)b * 2 * planeHW + planeHW +  (size_t)gy * W + gx] = acc1;
  }
}

// ---------------------------------------------------------------------------
extern "C" void kernel_launch(void* const* d_in, const int* in_sizes, int n_in,
                              void* d_out, int out_size, void* d_ws, size_t ws_size,
                              hipStream_t stream) {
  (void)in_sizes; (void)n_in; (void)out_size; (void)d_ws; (void)ws_size;
  const float* flow = (const float*)d_in[0];   // [2,2,1024,1024]
  const float* w1   = (const float*)d_in[1];   // [32,3,3,3]
  const float* b1   = (const float*)d_in[2];   // [32]
  const float* w2   = (const float*)d_in[3];   // [50,32,1,1]
  const float* b2   = (const float*)d_in[4];   // [50]
  float* out = (float*)d_out;                  // [2,2,1024,1024]

  const int H = 1024, W = 1024, B = 2;
  dim3 grid(W / 16, H / 16, B);
  spatial_refine_fused<<<grid, 256, 0, stream>>>(flow, w1, b1, w2, b2, out, H, W);
}
